// OlmoLMHeadCE_19765439496675
// MI455X (gfx1250) — compile-verified, matrix operand order checked by reading, not run on hardware
//
#include <hip/hip_runtime.h>

#define N_TOK 8192
#define HDIM  2048
#define VOCAB 50257
#define IGNORE_INDEX (-100)

#define ROWS_PER_BLOCK 128
#define CHUNK_COLS     2048
#define NCHUNK         ((VOCAB + CHUNK_COLS - 1) / CHUNK_COLS)   /* 25 */
#define COL_TILE       128
#define KSLAB          64
#define NSLAB          (HDIM / KSLAB)                            /* 32 */
#define LDS_STRIDE     72   /* ushort elems per row: 64 data + 8 pad (bank spread) */
#define TILE_USH       (ROWS_PER_BLOCK * LDS_STRIDE)

typedef __attribute__((ext_vector_type(16))) __bf16        v16bf;
typedef __attribute__((ext_vector_type(8)))  float         v8f;
typedef __attribute__((ext_vector_type(4)))  unsigned int  v4u;

union FragU { v4u u4[2]; v16bf bf; };

__device__ __forceinline__ unsigned short f2bf(float f) {
    union { float f; unsigned int u; } c; c.f = f;
    return (unsigned short)(c.u >> 16);
}

// CDNA5 16-bit 16x32 A/B fragment layout (wave32):
// lane m=lane&15 selects row; half=lane>>4 selects K sub-range.
// seg0: K = koff + half*8 .. +7 ; seg1: K = koff + 16 + half*8 .. +7
__device__ __forceinline__ v16bf load_frag(const unsigned short* tile_row0,
                                           int m, int half, int koff) {
    FragU fr;
    const unsigned short* p = tile_row0 + m * LDS_STRIDE + koff + half * 8;
    fr.u4[0] = *(const v4u*)(p);
    fr.u4[1] = *(const v4u*)(p + 16);
    return fr.bf;
}

// Issue global loads for one 128x64 f32 slab of x and W into registers.
__device__ __forceinline__ void stage_loads(const float* __restrict__ x,
                                            const float* __restrict__ w,
                                            int rowbase, int colblock, int kbase,
                                            int tid, float4 ax[8], float4 bx[8]) {
#pragma unroll
    for (int i = 0; i < 8; ++i) {
        const int idx = tid + i * 256;       // float4 index, 2048 total
        const int row = idx >> 4;
        const int k4  = idx & 15;
        ax[i] = *(const float4*)(x + (size_t)(rowbase + row) * HDIM + kbase + k4 * 4);
        const int vr = colblock + row;
        if (vr < VOCAB) bx[i] = *(const float4*)(w + (size_t)vr * HDIM + kbase + k4 * 4);
        else { bx[i].x = 0.f; bx[i].y = 0.f; bx[i].z = 0.f; bx[i].w = 0.f; }
    }
}

// Pack f32 -> bf16 and store staged slab into an LDS buffer.
__device__ __forceinline__ void stage_store(unsigned short* __restrict__ A,
                                            unsigned short* __restrict__ B,
                                            int tid, const float4 ax[8], const float4 bx[8]) {
#pragma unroll
    for (int i = 0; i < 8; ++i) {
        const int idx = tid + i * 256;
        const int row = idx >> 4;
        const int k4  = idx & 15;
        {
            const unsigned int u0 = ((unsigned int)f2bf(ax[i].y) << 16) | f2bf(ax[i].x);
            const unsigned int u1 = ((unsigned int)f2bf(ax[i].w) << 16) | f2bf(ax[i].z);
            *(uint2*)(&A[row * LDS_STRIDE + k4 * 4]) = make_uint2(u0, u1);
        }
        {
            const unsigned int u0 = ((unsigned int)f2bf(bx[i].y) << 16) | f2bf(bx[i].x);
            const unsigned int u1 = ((unsigned int)f2bf(bx[i].w) << 16) | f2bf(bx[i].z);
            *(uint2*)(&B[row * LDS_STRIDE + k4 * 4]) = make_uint2(u0, u1);
        }
    }
}

// 16 WMMAs for one K-slab, with register-double-buffered B fragments so the
// next fragment's ds_loads overlap the current WMMA.
__device__ __forceinline__ void compute_slab(const unsigned short* __restrict__ A,
                                             const unsigned short* __restrict__ B,
                                             int wave, int m, int half, v8f acc[8]) {
    v16bf af[2];
    af[0] = load_frag(A + wave * 16 * LDS_STRIDE, m, half, 0);
    af[1] = load_frag(A + wave * 16 * LDS_STRIDE, m, half, 32);
    v16bf bf2[2];
    bf2[0] = load_frag(B, m, half, 0);
#pragma unroll
    for (int s = 0; s < 16; ++s) {
        const int kk = s >> 3;
        const int t  = s & 7;
        if (s < 15) {
            const int s2  = s + 1;
            bf2[s2 & 1] = load_frag(B + (s2 & 7) * 16 * LDS_STRIDE, m, half, (s2 >> 3) * 32);
        }
        acc[t] = __builtin_amdgcn_wmma_f32_16x16x32_bf16(
            false, af[kk], false, bf2[s & 1], (short)0, acc[t], false, false);
    }
}

__global__ __launch_bounds__(256) void ce_partial_kernel(
    const float* __restrict__ x, const float* __restrict__ w,
    const int* __restrict__ target,
    float* __restrict__ ws_m, float* __restrict__ ws_s, float* __restrict__ ws_t)
{
    __shared__ unsigned short As[2][TILE_USH];
    __shared__ unsigned short Bs[2][TILE_USH];

    const int tid  = threadIdx.x;
    const int lane = tid & 31;
    const int wave = tid >> 5;        // 0..7 waves, 16 rows each
    const int m    = lane & 15;
    const int half = lane >> 4;

    const int rowbase   = blockIdx.x * ROWS_PER_BLOCK;
    const int chunkbase = blockIdx.y * CHUNK_COLS;

    // per-lane replicated online-softmax state for 8 rows (row = base + r + 8*half)
    float mstate[8], sstate[8], tacc[8];
    int   tcol[8];
#pragma unroll
    for (int r = 0; r < 8; ++r) {
        mstate[r] = -__builtin_inff(); sstate[r] = 0.f; tacc[r] = 0.f;
        const int row = rowbase + wave * 16 + r + 8 * half;
        tcol[r] = target[row];
    }

    for (int ct = 0; ct < CHUNK_COLS / COL_TILE; ++ct) {
        const int colblock = chunkbase + ct * COL_TILE;
        if (colblock >= VOCAB) break;

        v8f acc[8];
#pragma unroll
        for (int t = 0; t < 8; ++t)
#pragma unroll
            for (int e = 0; e < 8; ++e) acc[t][e] = 0.f;

        float4 ax[8], bx[8];

        // prologue: stage slab 0 into buffer 0
        stage_loads(x, w, rowbase, colblock, 0, tid, ax, bx);
        stage_store(As[0], Bs[0], tid, ax, bx);
        __syncthreads();

        for (int ks = 0; ks < NSLAB; ++ks) {
            const int cur  = ks & 1;
            const int more = (ks + 1 < NSLAB);

            // issue global loads for slab ks+1 (latency hides behind WMMAs)
            if (more)
                stage_loads(x, w, rowbase, colblock, (ks + 1) * KSLAB, tid, ax, bx);

            compute_slab(As[cur], Bs[cur], wave, m, half, acc);

            // pack + store slab ks+1 into the other buffer
            if (more)
                stage_store(As[cur ^ 1], Bs[cur ^ 1], tid, ax, bx);

            __syncthreads();
        }

        // ---- epilogue: online softmax over this 128-col tile ----
#pragma unroll
        for (int r = 0; r < 8; ++r) {
            float vals[8];
            float tmax = -__builtin_inff();
#pragma unroll
            for (int t = 0; t < 8; ++t) {
                const int col = colblock + t * 16 + m;
                const float vraw = acc[t][r];
                const float v = (col < VOCAB) ? vraw : -__builtin_inff();
                vals[t] = v;
                tmax = fmaxf(tmax, v);
                if (col == tcol[r]) tacc[r] += vraw;
            }
#pragma unroll
            for (int off = 8; off > 0; off >>= 1)
                tmax = fmaxf(tmax, __shfl_xor(tmax, off, 16));
            float se = 0.f;
#pragma unroll
            for (int t = 0; t < 8; ++t) se += __expf(vals[t] - tmax);
#pragma unroll
            for (int off = 8; off > 0; off >>= 1)
                se += __shfl_xor(se, off, 16);
            const float nm = fmaxf(mstate[r], tmax);
            sstate[r] = sstate[r] * __expf(mstate[r] - nm) + se * __expf(tmax - nm);
            mstate[r] = nm;
        }
    }

    // reduce target contributions across the 16-lane group
#pragma unroll
    for (int r = 0; r < 8; ++r)
#pragma unroll
        for (int off = 8; off > 0; off >>= 1)
            tacc[r] += __shfl_xor(tacc[r], off, 16);

    if ((lane & 15) == 0) {
#pragma unroll
        for (int r = 0; r < 8; ++r) {
            const int row = rowbase + wave * 16 + r + 8 * half;
            const size_t o = (size_t)row * NCHUNK + blockIdx.y;
            ws_m[o] = mstate[r];
            ws_s[o] = sstate[r];
            ws_t[o] = tacc[r];
        }
    }
}

__global__ __launch_bounds__(256) void ce_rowreduce_kernel(
    const float* __restrict__ ws_m, const float* __restrict__ ws_s,
    const float* __restrict__ ws_t, const int* __restrict__ target,
    float* __restrict__ ws_loss)
{
    const int row = blockIdx.x * blockDim.x + threadIdx.x;
    if (row >= N_TOK) return;
    float M = -__builtin_inff();
    for (int c = 0; c < NCHUNK; ++c)
        M = fmaxf(M, ws_m[(size_t)row * NCHUNK + c]);
    float S = 0.f, T = 0.f;
    for (int c = 0; c < NCHUNK; ++c) {
        S += ws_s[(size_t)row * NCHUNK + c] * __expf(ws_m[(size_t)row * NCHUNK + c] - M);
        T += ws_t[(size_t)row * NCHUNK + c];
    }
    const float lse = M + __logf(S);
    ws_loss[row] = (target[row] != IGNORE_INDEX) ? (lse - T) : 0.f;
}

__global__ __launch_bounds__(256) void ce_final_reduce(
    const float* __restrict__ ws_loss, float* __restrict__ out)
{
    __shared__ float red[256];
    float s = 0.f;
    for (int i = threadIdx.x; i < N_TOK; i += 256) s += ws_loss[i];
    red[threadIdx.x] = s;
    __syncthreads();
    for (int off = 128; off > 0; off >>= 1) {
        if ((int)threadIdx.x < off) red[threadIdx.x] += red[threadIdx.x + off];
        __syncthreads();
    }
    if (threadIdx.x == 0) out[0] = red[0];
}

extern "C" void kernel_launch(void* const* d_in, const int* in_sizes, int n_in,
                              void* d_out, int out_size, void* d_ws, size_t ws_size,
                              hipStream_t stream)
{
    const float* x      = (const float*)d_in[0];
    const float* w      = (const float*)d_in[1];
    const int*   target = (const int*)d_in[2];

    float* ws      = (float*)d_ws;
    float* ws_m    = ws;
    float* ws_s    = ws_m + (size_t)N_TOK * NCHUNK;
    float* ws_t    = ws_s + (size_t)N_TOK * NCHUNK;
    float* ws_loss = ws_t + (size_t)N_TOK * NCHUNK;

    dim3 grid(N_TOK / ROWS_PER_BLOCK, NCHUNK);   // chunk = slow index -> L2 reuse of W chunk
    ce_partial_kernel<<<grid, 256, 0, stream>>>(x, w, target, ws_m, ws_s, ws_t);
    ce_rowreduce_kernel<<<(N_TOK + 255) / 256, 256, 0, stream>>>(ws_m, ws_s, ws_t, target, ws_loss);
    ce_final_reduce<<<1, 256, 0, stream>>>(ws_loss, (float*)d_out);
}
